// MultiShiftedWindow_36558761623691
// MI455X (gfx1250) — matrix operand order repaired and checked by reading, not compiled, not run on hardware
//
#include <hip/hip_runtime.h>

// ---------------------------------------------------------------------------
// CDNA5 (gfx1250) WMMA building blocks
// ---------------------------------------------------------------------------
typedef _Float16 v16h __attribute__((ext_vector_type(16)));
typedef float    v8f  __attribute__((ext_vector_type(8)));

#define LSTR 36   // LDS row stride (f16) for 32-wide K tiles; 18-bank step => conflict free

__device__ __forceinline__ v8f wmma_f16(v16h a, v16h b, v8f c) {
  // D = A(16x32,f16) * B(32x16,f16) + C(16x16,f32)
  return __builtin_amdgcn_wmma_f32_16x16x32_f16(false, a, false, b, (short)0, c,
                                                false, false);
}

// CDNA5 async DMA: global -> LDS, 8 bytes per lane, tracked by ASYNCcnt.
// GV mode: vdst = per-lane LDS byte address, v[addr:addr+1] = 64-bit address.
__device__ __forceinline__ void async_load_b64(unsigned lds_off, const void* gaddr) {
  asm volatile("global_load_async_to_lds_b64 %0, %1, off"
               :: "v"(lds_off), "v"(gaddr) : "memory");
}
__device__ __forceinline__ void wait_async0() {
  asm volatile("s_wait_asynccnt 0x0" ::: "memory");
}

// A fragment: 16(M) x 32(K) f16 from a row-major [M][stride] tile.
// Per ISA 7.12.2: lane(0-15)=row M, half selects K blocks of 8; 2 f16 per VGPR.
__device__ __forceinline__ v16h load_frag_a(const _Float16* p, int stride) {
  const int lane = threadIdx.x & 31;
  const int m = lane & 15, hi = lane >> 4;
  union { v16h h; unsigned u[8]; } t;
#pragma unroll
  for (int v = 0; v < 8; ++v) {
    const int k = (v < 4) ? (hi * 8 + v * 2) : (16 + hi * 8 + (v - 4) * 2);
    t.u[v] = *reinterpret_cast<const unsigned*>(p + m * stride + k);
  }
  return t.h;
}

// B fragment: 32(K) x 16(N), memory holds B^T as [N][stride] row-major
// (weight layout [Nout][Kin]).  Same K order as A: half selects K blocks of 8.
__device__ __forceinline__ v16h load_frag_b(const _Float16* p, int stride) {
  const int lane = threadIdx.x & 31;
  const int n = lane & 15, hi = lane >> 4;
  union { v16h h; unsigned u[8]; } t;
#pragma unroll
  for (int v = 0; v < 8; ++v) {
    const int k = (v < 4) ? (hi * 8 + v * 2) : (16 + hi * 8 + (v - 4) * 2);
    t.u[v] = *reinterpret_cast<const unsigned*>(p + n * stride + k);
  }
  return t.h;
}

// ---------------------------------------------------------------------------
// f32 -> f16 elementwise convert (weights, once per launch; graph safe)
// ---------------------------------------------------------------------------
__global__ void cvt_f32_f16_kernel(const float* __restrict__ in,
                                   _Float16* __restrict__ out, size_t n) {
  size_t i = (size_t)blockIdx.x * blockDim.x + threadIdx.x;
  const size_t st = (size_t)gridDim.x * blockDim.x;
  for (; i < n; i += st) out[i] = (_Float16)in[i];
}

// ---------------------------------------------------------------------------
// LayerNorm over C=512, one block (256 thr) per row, f16 output
// ---------------------------------------------------------------------------
__launch_bounds__(256)
__global__ void layernorm512_kernel(const float* __restrict__ x,
                                    const float* __restrict__ g,
                                    const float* __restrict__ b,
                                    _Float16* __restrict__ out) {
  __shared__ float red[256];
  const int C = 512;
  const int t = threadIdx.x;
  const size_t row = blockIdx.x;
  const float* xr = x + row * C;
  float v0 = xr[t], v1 = xr[t + 256];
  red[t] = v0 + v1;
  __syncthreads();
  for (int s = 128; s > 0; s >>= 1) {
    if (t < s) red[t] += red[t + s];
    __syncthreads();
  }
  const float mu = red[0] * (1.0f / C);
  __syncthreads();
  const float d0 = v0 - mu, d1 = v1 - mu;
  red[t] = d0 * d0 + d1 * d1;
  __syncthreads();
  for (int s = 128; s > 0; s >>= 1) {
    if (t < s) red[t] += red[t + s];
    __syncthreads();
  }
  const float rstd = rsqrtf(red[0] * (1.0f / C) + 1e-5f);
  out[row * C + t]       = (_Float16)(d0 * rstd * g[t] + b[t]);
  out[row * C + t + 256] = (_Float16)(d1 * rstd * g[t + 256] + b[t + 256]);
}

// ---------------------------------------------------------------------------
// Generic WMMA GEMM:  out = epi( A[M,K](f16) @ B[N,K]^T(f16) )
// Block: 256 thr (8 waves), tile 128(M) x 128(N); wave tile 32x64
// => 8 v_wmma per k-step against 6 LDS fragment loads.
// Tiles staged with global_load_async_to_lds_b64 (ASYNCcnt, no VGPR hop).
// flags: bit0 = ReLU, bit1 = accumulate into existing f32 C
// epi order: +bias -> *scalev -> relu -> +residual -> (+C old) -> store
// Requires M%128==0, N%128==0, K%32==0 (true for all uses here).
// ---------------------------------------------------------------------------
__launch_bounds__(256)
__global__ void gemm_wmma_kernel(const _Float16* __restrict__ A, int lda,
                                 const _Float16* __restrict__ B, int ldb,
                                 float* __restrict__ C, int ldc,
                                 _Float16* __restrict__ outH, int ldo,
                                 const float* __restrict__ bias,
                                 const float* __restrict__ residual, int ldr,
                                 const float* __restrict__ scalev,
                                 int K, int flags) {
  __shared__ _Float16 sA[128 * LSTR];
  __shared__ _Float16 sB[128 * LSTR];
  const int tid = threadIdx.x;
  const int wv = tid >> 5;
  const int wm = wv >> 1;            // 0..3  (32 rows each)
  const int wn = wv & 1;             // 0..1  (64 cols each)
  const int bm = blockIdx.x * 128;
  const int bn = blockIdx.y * 128;

  v8f acc[2][4] = {};

  for (int k0 = 0; k0 < K; k0 += 32) {
    __syncthreads();   // previous tile fully consumed
    // stage A and B tiles (each 128x32 f16 = 1024 8-byte chunks) via async DMA
#pragma unroll
    for (int i = 0; i < 4; ++i) {
      const int id = tid + i * 256;
      const int row = id >> 3, c4 = (id & 7) * 4;
      async_load_b64((unsigned)(size_t)&sA[row * LSTR + c4],
                     &A[(size_t)(bm + row) * lda + k0 + c4]);
      async_load_b64((unsigned)(size_t)&sB[row * LSTR + c4],
                     &B[(size_t)(bn + row) * ldb + k0 + c4]);
    }
    // pull the next K-tile toward L2/WGP$ (global_prefetch_b8)
    if (k0 + 32 < K) {
      const int prow = tid >> 1;
      __builtin_prefetch(&A[(size_t)(bm + prow) * lda + k0 + 32], 0, 3);
      __builtin_prefetch(&B[(size_t)(bn + prow) * ldb + k0 + 32], 0, 3);
    }
    wait_async0();     // this wave's DMA landed in LDS
    __syncthreads();   // all waves' DMA landed
    const v16h a0 = load_frag_a(&sA[(wm * 32) * LSTR], LSTR);
    const v16h a1 = load_frag_a(&sA[(wm * 32 + 16) * LSTR], LSTR);
#pragma unroll
    for (int ni = 0; ni < 4; ++ni) {
      const v16h bf = load_frag_b(&sB[(wn * 64 + ni * 16) * LSTR], LSTR);
      acc[0][ni] = wmma_f16(a0, bf, acc[0][ni]);
      acc[1][ni] = wmma_f16(a1, bf, acc[1][ni]);
    }
  }

  // epilogue: C/D layout — lane(0-15)=col, half selects row block of 8
  const int lane = tid & 31;
  const int cn = lane & 15;
  const int rb = (lane >> 4) * 8;
#pragma unroll
  for (int mi = 0; mi < 2; ++mi)
#pragma unroll
    for (int ni = 0; ni < 4; ++ni) {
      const int gn = bn + wn * 64 + ni * 16 + cn;
      const float bv = bias ? bias[gn] : 0.0f;
      const float sv = scalev ? scalev[gn] : 1.0f;
#pragma unroll
      for (int r = 0; r < 8; ++r) {
        const int gm = bm + wm * 32 + mi * 16 + rb + r;
        float v = acc[mi][ni][r] + bv;
        v *= sv;
        if (flags & 1) v = fmaxf(v, 0.0f);
        if (residual) v += residual[(size_t)gm * ldr + gn];
        if (outH) {
          outH[(size_t)gm * ldo + gn] = (_Float16)v;
        } else {
          if (flags & 2) v += C[(size_t)gm * ldc + gn];
          C[(size_t)gm * ldc + gn] = v;
        }
      }
    }
}

// ---------------------------------------------------------------------------
// Shifted-window attention, one block = (window, head). 128 threads (4 waves).
// hd = 32 => S = Q.K^T is exactly ONE wmma K-step per 16x16 score tile.
// Shift mask computed analytically (Swin region ids); roll folded into the
// token index map.  qkv: [B*L][1536] f16 (Q|K|V), out: [B*L][512] f16.
// ---------------------------------------------------------------------------
__device__ __forceinline__ int swin_reg(int p, int w, int shift) {
  if (p >= 56 - shift) return 2;
  if (p >= 56 - w) return 1;
  return 0;
}

__launch_bounds__(128)
__global__ void window_attn_kernel(const _Float16* __restrict__ qkv,
                                   _Float16* __restrict__ outh,
                                   int w, int shift) {
  extern __shared__ _Float16 smem[];
  const int H = 56, L = 3136, C3 = 1536, CO = 512, HD = 32;
  const int nw = H / w;
  const int ww = w * w;
  const int KP = (ww + 31) & ~31;     // padded contraction dim for P@V
  const int MT = (ww + 15) >> 4;      // 16-row tiles (== 16-col score tiles)
  const int NP = MT * 16;

  _Float16* sQ  = smem;               // [NP][LSTR]
  _Float16* sK  = sQ + NP * LSTR;     // [NP][LSTR]
  _Float16* sVt = sK + NP * LSTR;     // [32][KP]   (V transposed: [hd][tok])
  _Float16* sP  = sVt + 32 * KP;      // [16][KP]
  float*    sS  = reinterpret_cast<float*>(sP + 16 * KP); // [16][NP]

  const int widx = blockIdx.x;
  const int b = widx / (nw * nw);
  const int wrem = widx % (nw * nw);
  const int wi = wrem / nw, wj = wrem % nw;
  const int head = blockIdx.y;
  const int tid = threadIdx.x;
  const int lane = tid & 31, wv4 = tid >> 5;

  // zero Q/K/Vt/P regions (covers all padding)
  const int zn = NP * LSTR * 2 + 32 * KP + 16 * KP;
  for (int i = tid; i < zn; i += 128) smem[i] = (_Float16)0;
  __syncthreads();

  // token map: fold jnp.roll(-shift) + window partition into one index
  auto tok = [&](int q) -> int {
    const int r = q / w, c = q % w;
    int hh = wi * w + r + shift; if (hh >= H) hh -= H;
    int cc = wj * w + c + shift; if (cc >= H) cc -= H;
    return b * L + hh * H + cc;
  };

  // stage Q and K rows via async gather-DMA (8-byte chunks), V transposed scalar
  for (int i = tid; i < ww * 8; i += 128) {
    const int row = i >> 3, c4 = (i & 7) * 4;
    const _Float16* src = qkv + (size_t)tok(row) * C3 + head * HD;
    async_load_b64((unsigned)(size_t)&sQ[row * LSTR + c4], src + c4);
    async_load_b64((unsigned)(size_t)&sK[row * LSTR + c4], src + 512 + c4);
  }
  for (int i = tid; i < ww * HD; i += 128) {
    const int row = i >> 5, d = i & 31;
    sVt[d * KP + row] = qkv[(size_t)tok(row) * C3 + 1024 + head * HD + d];
  }
  wait_async0();
  __syncthreads();

  const float scale = 0.17677669529663687f;  // 32^-0.5

  for (int mt = 0; mt < MT; ++mt) {
    // ---- S = Q.K^T for this 16-row strip (one wmma per 16x16 tile) ----
    const v16h aq = load_frag_a(&sQ[mt * 16 * LSTR], LSTR);
    for (int nt = wv4; nt < MT; nt += 4) {
      const v16h bk = load_frag_b(&sK[nt * 16 * LSTR], LSTR);
      v8f s = {};
      s = wmma_f16(aq, bk, s);
      const int cn = lane & 15, rb = (lane >> 4) * 8;
#pragma unroll
      for (int r = 0; r < 8; ++r) {
        const int qi = mt * 16 + rb + r;
        const int ki = nt * 16 + cn;
        float v = s[r] * scale;
        if (qi >= ww || ki >= ww) {
          v = -1e30f;
        } else if (shift > 0) {
          const int rq = swin_reg(wi * w + qi / w, w, shift) * 3 +
                         swin_reg(wj * w + qi % w, w, shift);
          const int rk = swin_reg(wi * w + ki / w, w, shift) * 3 +
                         swin_reg(wj * w + ki % w, w, shift);
          if (rq != rk) v += -100.0f;
        }
        sS[(rb + r) * NP + nt * 16 + cn] = v;
      }
    }
    __syncthreads();

    // ---- softmax: 16 rows x 8 threads (in-wave shuffles) ----
    {
      const int row = tid >> 3, sub = tid & 7;
      float mx = -1e30f;
      for (int k = sub; k < ww; k += 8) mx = fmaxf(mx, sS[row * NP + k]);
      for (int o = 4; o > 0; o >>= 1) mx = fmaxf(mx, __shfl_xor(mx, o, 8));
      float sum = 0.0f;
      for (int k = sub; k < ww; k += 8) sum += __expf(sS[row * NP + k] - mx);
      for (int o = 4; o > 0; o >>= 1) sum += __shfl_xor(sum, o, 8);
      const float inv = 1.0f / sum;
      for (int k = sub; k < KP; k += 8)
        sP[row * KP + k] =
            (k < ww) ? (_Float16)(__expf(sS[row * NP + k] - mx) * inv)
                     : (_Float16)0;
    }
    __syncthreads();

    // ---- O = P @ V : two 16-col (hd) tiles, K stepped by 32 ----
    if (wv4 < 2) {
      v8f o = {};
      for (int ks = 0; ks < KP; ks += 32) {
        const v16h ap = load_frag_a(&sP[ks], KP);
        const v16h bv = load_frag_b(&sVt[wv4 * 16 * KP + ks], KP);
        o = wmma_f16(ap, bv, o);
      }
      const int cn = lane & 15, rb = (lane >> 4) * 8;
#pragma unroll
      for (int r = 0; r < 8; ++r) {
        const int qi = mt * 16 + rb + r;
        if (qi < ww)
          outh[(size_t)tok(qi) * CO + head * HD + wv4 * 16 + cn] = (_Float16)o[r];
      }
    }
    __syncthreads();
  }
}

// ---------------------------------------------------------------------------
// Host-side orchestration
// ---------------------------------------------------------------------------
extern "C" void kernel_launch(void* const* d_in, const int* in_sizes, int n_in,
                              void* d_out, int out_size, void* d_ws, size_t ws_size,
                              hipStream_t stream) {
  (void)in_sizes; (void)n_in; (void)out_size; (void)ws_size;
  const float* x        = (const float*)d_in[0];
  const float* qkv_w    = (const float*)d_in[1];
  const float* qkv_b    = (const float*)d_in[2];
  const float* proj_w   = (const float*)d_in[3];
  const float* proj_b   = (const float*)d_in[4];
  const float* n1g      = (const float*)d_in[5];
  const float* n1b      = (const float*)d_in[6];
  const float* n2g      = (const float*)d_in[7];
  const float* n2b      = (const float*)d_in[8];
  const float* reduce_w = (const float*)d_in[9];
  const float* ffn_w1   = (const float*)d_in[10];
  const float* ffn_b1   = (const float*)d_in[11];
  const float* ffn_w2   = (const float*)d_in[12];
  const float* ffn_b2   = (const float*)d_in[13];
  const float* ls       = (const float*)d_in[14];
  float* out = (float*)d_out;

  const int Bz = 8, C = 512, L = 3136, F = 2048;
  const int M = Bz * L;                 // 25088 (divisible by 128)

  // deterministic workspace carve (256B aligned)
  char* ws = (char*)d_ws;
  size_t off = 0;
  auto carve = [&](size_t bytes) -> void* {
    void* p = ws + off;
    off = (off + bytes + 255) & ~(size_t)255;
    return p;
  };
  _Float16* wq16  = (_Float16*)carve(sizeof(_Float16) * (size_t)8 * 1536 * 512);
  _Float16* wp16  = (_Float16*)carve(sizeof(_Float16) * (size_t)8 * 512 * 512);
  _Float16* wr16  = (_Float16*)carve(sizeof(_Float16) * (size_t)512 * 4096);
  _Float16* w116  = (_Float16*)carve(sizeof(_Float16) * (size_t)F * C);
  _Float16* w216  = (_Float16*)carve(sizeof(_Float16) * (size_t)C * F);
  _Float16* xn16  = (_Float16*)carve(sizeof(_Float16) * (size_t)M * C);
  _Float16* qkv16 = (_Float16*)carve(sizeof(_Float16) * (size_t)M * 1536);
  _Float16* att16 = (_Float16*)carve(sizeof(_Float16) * (size_t)M * C);
  _Float16* br16  = (_Float16*)carve(sizeof(_Float16) * (size_t)M * C);
  float*    xr32  = (float*)   carve(sizeof(float)    * (size_t)M * C);
  _Float16* xrn16 = (_Float16*)carve(sizeof(_Float16) * (size_t)M * C);
  _Float16* h116  = (_Float16*)carve(sizeof(_Float16) * (size_t)M * F);

  // 1) weights -> f16
  auto cvt = [&](const float* src, _Float16* dst, size_t n) {
    cvt_f32_f16_kernel<<<2048, 256, 0, stream>>>(src, dst, n);
  };
  cvt(qkv_w,    wq16, (size_t)8 * 1536 * 512);
  cvt(proj_w,   wp16, (size_t)8 * 512 * 512);
  cvt(reduce_w, wr16, (size_t)512 * 4096);
  cvt(ffn_w1,   w116, (size_t)F * C);
  cvt(ffn_w2,   w216, (size_t)C * F);

  // 2) LN1 -> xn16
  layernorm512_kernel<<<M, 256, 0, stream>>>(x, n1g, n1b, xn16);

  auto gemm = [&](const _Float16* A, int lda, const _Float16* B, int ldb,
                  float* Cp, int ldc, _Float16* oH, int ldo,
                  const float* bias, const float* res, int ldr,
                  const float* sc, int N, int K, int flags) {
    dim3 g(M / 128, N / 128);
    gemm_wmma_kernel<<<g, 256, 0, stream>>>(A, lda, B, ldb, Cp, ldc, oH, ldo,
                                            bias, res, ldr, sc, K, flags);
  };

  // 3) eight shifted-window branches; xr accumulates Σ_i branch_i @ Wr_i^T
  const int wins[8]   = {4, 4, 7, 7, 8, 8, 14, 14};
  const int shifts[8] = {0, 2, 0, 3, 0, 4, 0, 7};
  for (int i = 0; i < 8; ++i) {
    const int w = wins[i], shift = shifts[i];
    // QKV: [M,512] @ [1536,512]^T -> qkv16 [M,1536]
    gemm(xn16, C, wq16 + (size_t)i * 1536 * 512, C, nullptr, 0, qkv16, 1536,
         qkv_b + (size_t)i * 1536, nullptr, 0, nullptr, 1536, C, 0);
    // attention
    const int nw = 56 / w, ww = w * w;
    const int KP = (ww + 31) & ~31, MT = (ww + 15) >> 4, NP = MT * 16;
    const size_t smem =
        (size_t)(NP * LSTR * 2 + 32 * KP + 16 * KP) * sizeof(_Float16) +
        (size_t)16 * NP * sizeof(float);
    dim3 ag(Bz * nw * nw, 16);
    window_attn_kernel<<<ag, 128, smem, stream>>>(qkv16, att16, w, shift);
    // proj + residual(x):  br16 = att16 @ pw^T + pb + x
    gemm(att16, C, wp16 + (size_t)i * 512 * 512, C, nullptr, 0, br16, C,
         proj_b + (size_t)i * 512, x, C, nullptr, C, C, 0);
    // reduce accumulate:   xr32 (+)= br16 @ Wr_i^T   (Wr_i = reduce_w[:, i*512:])
    gemm(br16, C, wr16 + (size_t)i * 512, 4096, xr32, C, nullptr, 0,
         nullptr, nullptr, 0, nullptr, C, C, (i > 0) ? 2 : 0);
  }

  // 4) LN2 -> xrn16
  layernorm512_kernel<<<M, 256, 0, stream>>>(xr32, n2g, n2b, xrn16);

  // 5) FFN1: h116 = relu(xrn16 @ w1^T + b1)
  gemm(xrn16, C, w116, C, nullptr, 0, h116, F, ffn_b1, nullptr, 0, nullptr,
       F, C, 1);

  // 6) FFN2 + LayerScale + residual: out = xr32 + (h116 @ w2^T + b2) * ls
  gemm(h116, F, w216, F, out, C, nullptr, 0, ffn_b2, xr32, C, ls, C, F, 0);
}
